// SeMCA_74655121539296
// MI455X (gfx1250) — matrix operand order repaired and checked by reading, not compiled, not run on hardware
//
#include <hip/hip_runtime.h>

typedef __attribute__((ext_vector_type(16))) _Float16 v16h;
typedef __attribute__((ext_vector_type(8)))  _Float16 v8h;
typedef __attribute__((ext_vector_type(8)))  float    v8f;

#define CC   64
#define IMGN 21
#define DD   441

// ---------------------------------------------------------------------------
// Kernel A: k1 (depthwise conv1d over channel axis == banded 64x64 correlation
// per spatial position) and v (depthwise 3x3 conv). Both independent.
//   blocks [0,441):   d = blockIdx.x, 128 threads = (b in 2) x (c in 64)
//   blocks [441,569): (b,c) pair, 128 threads loop 441 pixels
// k1_ws layout: [b][d][c]   (d = i*21+j),   v_ws layout: [b][c][i*21+j]
// ---------------------------------------------------------------------------
__global__ __launch_bounds__(128) void semca_stage1(
    const float* __restrict__ x,  const float* __restrict__ wk,
    const float* __restrict__ bk, const float* __restrict__ wv,
    const float* __restrict__ bv, float* __restrict__ k1_ws,
    float* __restrict__ v_ws) {
  __shared__ float xv[2][64];
  __shared__ float wkd[128];   // wk[d, 157..283] (127 taps actually used)
  const int t = threadIdx.x;
  if (blockIdx.x < DD) {
    const int d = blockIdx.x;
    {
      int b = t >> 6, c = t & 63;
      xv[b][c] = x[(b * CC + c) * DD + d];
    }
    if (t < 127) wkd[t] = wk[d * DD + 157 + t];
    __syncthreads();
    const int b = t >> 6, c = t & 63;
    // k1[b,d,c] = sum_{c'} wk[d, 220 + c' - c] * x[b,c',d] + bk[d]
    float acc = bk[d];
#pragma unroll 8
    for (int cp = 0; cp < 64; ++cp)
      acc += wkd[63 + cp - c] * xv[b][cp];   // 63 == 220-157
    k1_ws[(b * DD + d) * CC + c] = acc;
  } else {
    const int idx = blockIdx.x - DD;
    const int b = idx >> 6, c = idx & 63;
    float ww[9];
#pragma unroll
    for (int q = 0; q < 9; ++q) ww[q] = wv[c * 9 + q];
    const float bb = bv[c];
    const float* xp = x + (size_t)(b * CC + c) * DD;
    for (int p = t; p < DD; p += 128) {
      int i = p / IMGN, j = p - i * IMGN;
      float acc = bb;
#pragma unroll
      for (int di = 0; di < 3; ++di) {
        int r = i + di - 1;
        if (r < 0 || r >= IMGN) continue;
#pragma unroll
        for (int dj = 0; dj < 3; ++dj) {
          int cl = j + dj - 1;
          if (cl < 0 || cl >= IMGN) continue;
          acc += xp[r * IMGN + cl] * ww[di * 3 + dj];
        }
      }
      v_ws[(b * CC + c) * DD + p] = acc;
    }
  }
}

// ---------------------------------------------------------------------------
// Kernel B: one workgroup per (b, g) with g = phase index ph*5+pw (25 groups).
// Pipeline (all in one WG, 256 threads = 8 waves):
//   0) stage kq^T (64x64 f16, n-major) + w1[g] (f16), BN params (zero-padded
//      to 32 so the writeback is branch-free), b2[g], v columns
//   1) att1 = w1[g] @ kq  via v_wmma_f32_16x16x32_f16; fused BN+ReLU,
//      written transposed (a1T[col][o]) with a single ds_store_b128 per lane
//   2) att2 = w2[g] @ att1 + b2[g] -> 441x64 f32 in LDS (A-frags from global)
//   3) column softmax over 441 rows (max-sub, exp kept unnormalized, 1/sum)
//   4) out[b,c,y,x] = inv[c] * sum_c' vloc * exp_att2[220-c+c', c] + k1_pad
// B operands are stored n-major so every B-fragment is 16 contiguous halfs
// per lane (two ds_load_b128) instead of 16 strided ds_load_u16.
// ---------------------------------------------------------------------------
__global__ __launch_bounds__(256) void semca_attn(
    const float* __restrict__ x,  const float* __restrict__ w1,
    const float* __restrict__ bn_gamma, const float* __restrict__ bn_beta,
    const float* __restrict__ bn_mean,  const float* __restrict__ bn_var,
    const float* __restrict__ w2, const float* __restrict__ b2,
    const float* __restrict__ k1_ws, const float* __restrict__ v_ws,
    float* __restrict__ out) {
  __shared__ float    att2[DD * CC];                  // 112896 B
  __shared__ __align__(32) _Float16 kqT[64 * 64];     // kq^T: [n][k], k-pad 50->64
  __shared__ __align__(32) _Float16 w1h[32 * 64];     // w1[g]: [o][k] row-major
  __shared__ __align__(32) _Float16 a1T[64 * 32];     // att1^T: [col][o], o-pad->32
  __shared__ float    vloc[25 * CC];
  __shared__ float    b2s[DD];
  __shared__ __align__(16) float bnm_s[32], bni_s[32], bnb_s[32];
  __shared__ float    red[256], red2[256], colinv[64];

  const int tid = threadIdx.x;
  const int wg  = blockIdx.x;
  const int b   = wg / 25, g = wg % 25;
  const int g1  = g / 5,  g2 = g % 5;
  // output positions owned by this g: (y+2)%5==g1, (x+2)%5==g2
  const int y0 = (g1 + 3) % 5, x0 = (g2 + 3) % 5;
  const int ny = (20 - y0) / 5 + 1, nx = (20 - x0) / 5 + 1;
  const int npos = ny * nx;

  __builtin_prefetch(w2 + (size_t)g * DD * 25, 0, 1);  // global_prefetch_b8

  // ---- phase 0: staging ----
  for (int i = tid; i < 32 * 64; i += 256) {
    int o = i >> 6, k = i & 63;
    float vv = (o < 25 && k < 50) ? w1[((size_t)g * 25 + o) * 50 + k] : 0.f;
    w1h[i] = (_Float16)vv;
  }
  // kq^T: i = n*64 + k so consecutive threads write consecutive LDS halfs
  for (int i = tid; i < 64 * 64; i += 256) {
    int v = i >> 6, ir = i & 63;   // v = channel-dim col (n), ir = kq row (k)
    float val = 0.f;
    if (ir < 50) {
      int u   = (ir >= 25) ? ir - 25 : ir;
      int mm  = u * 64 + v;                 // flat index re-factorized
      int cch = mm / 25, gg = mm - cch * 25;
      int r   = (gg / 5) * 5 + g1, cl = (gg % 5) * 5 + g2;
      if (r < IMGN && cl < IMGN) {
        val = (ir >= 25) ? k1_ws[((size_t)b * DD + r * IMGN + cl) * CC + cch]
                         : x[((size_t)b * CC + cch) * DD + r * IMGN + cl];
      }
    }
    kqT[i] = (_Float16)val;
  }
  for (int i = tid; i < DD; i += 256) b2s[i] = b2[(size_t)g * DD + i];
  if (tid < 32) {
    // rows 25..31 padded with (mean=0, inv=0, beta=0) so BN+ReLU yields 0
    const bool okk = tid < 25;
    bnm_s[tid] = okk ? bn_mean[g * 25 + tid] : 0.f;
    bni_s[tid] = okk ? bn_gamma[g * 25 + tid] * rsqrtf(bn_var[g * 25 + tid] + 1e-5f)
                     : 0.f;
    bnb_s[tid] = okk ? bn_beta[g * 25 + tid] : 0.f;
  }
  for (int i = tid; i < npos * CC; i += 256) {
    int p = i >> 6, cp = i & 63;
    int y = y0 + 5 * (p / nx), xx = x0 + 5 * (p % nx);
    vloc[i] = (y >= 2 && xx >= 2)
                  ? v_ws[((size_t)b * CC + cp) * DD + (y - 2) * IMGN + (xx - 2)]
                  : 0.f;
  }
  __syncthreads();

  const int lane = tid & 31;
  const int wid  = tid >> 5;       // 8 waves
  const int m    = lane & 15;      // row (A) / col (B,C) within tile
  const int hh   = lane >> 4;      // lane half

  // ---- phase 1: att1 = w1[g](25x50) @ kq(50x64), 8 tiles over 8 waves ----
  {
    const int mt = wid >> 2, nt = wid & 3;
    v8f acc = {0.f, 0.f, 0.f, 0.f, 0.f, 0.f, 0.f, 0.f};
#pragma unroll
    for (int kc = 0; kc < 2; ++kc) {
      // A frag: rows of w1h, two contiguous 16B halves per lane
      const int arow = (mt * 16 + m) * 64 + kc * 32;
      v8h alo = *(const v8h*)&w1h[arow + 16 * hh];
      v8h ahi = *(const v8h*)&w1h[arow + 16 * hh + 8];
      v16h af = __builtin_shufflevector(alo, ahi, 0, 1, 2, 3, 4, 5, 6, 7,
                                        8, 9, 10, 11, 12, 13, 14, 15);
      // B frag: contiguous 32B in kq^T (k runs fastest)
      v16h bf = *(const v16h*)&kqT[(nt * 16 + m) * 64 + kc * 32 + 16 * hh];
      acc = __builtin_amdgcn_wmma_f32_16x16x32_f16(false, af, false, bf,
                                                   (short)0, acc, false, false);
    }
    // fused BN (eval) + ReLU -> transposed f16 writeback (one b128 store).
    // Branch-free: padded BN params produce exact 0 for rows o in [25,32).
    const int o0 = mt * 16 + 8 * hh;   // 8 consecutive BN rows per lane
    v8h pack;
#pragma unroll
    for (int r = 0; r < 8; ++r) {
      float yv = (acc[r] - bnm_s[o0 + r]) * bni_s[o0 + r] + bnb_s[o0 + r];
      yv = yv > 0.f ? yv : 0.f;
      pack[r] = (_Float16)yv;
    }
    *(v8h*)&a1T[(nt * 16 + m) * 32 + o0] = pack;
  }
  __syncthreads();

  // ---- phase 2: att2 = w2[g](441x25) @ att1(25x64) + b2[g] ----
  {
    v16h bf[4];                                  // B frags reused by all d-tiles
#pragma unroll
    for (int nt = 0; nt < 4; ++nt)
      bf[nt] = *(const v16h*)&a1T[(nt * 16 + m) * 32 + 16 * hh];
    for (int dt = wid; dt < 28; dt += 8) {       // wave-uniform trip count
      const int  ddr  = dt * 16 + m;
      const bool rowok = ddr < DD;
      const float* wrow = w2 + ((size_t)g * DD + (rowok ? ddr : 0)) * 25;
      v16h af;
#pragma unroll
      for (int e = 0; e < 8; ++e) {              // cols 16h..16h+7 (<24, valid)
        float vv = rowok ? wrow[16 * hh + e] : 0.f;
        af[e] = (_Float16)vv;
      }
#pragma unroll
      for (int e = 8; e < 16; ++e) {             // cols 16h+8.. (valid if <25)
        int kk = 16 * hh + e;
        float vv = (rowok && kk < 25) ? wrow[kk] : 0.f;
        af[e] = (_Float16)vv;
      }
#pragma unroll
      for (int nt = 0; nt < 4; ++nt) {
        v8f acc = {0.f, 0.f, 0.f, 0.f, 0.f, 0.f, 0.f, 0.f};
        acc = __builtin_amdgcn_wmma_f32_16x16x32_f16(false, af, false, bf[nt],
                                                     (short)0, acc, false, false);
#pragma unroll
        for (int r = 0; r < 8; ++r) {
          int dd = dt * 16 + 8 * hh + r;
          if (dd < DD) att2[dd * 64 + nt * 16 + m] = acc[r] + b2s[dd];
        }
      }
    }
  }
  __syncthreads();

  // ---- phase 3: softmax over d (441) per column c; keep exp + 1/sum ----
  {
    const int c = tid & 63, part = tid >> 6;     // 4 partials per column
    const int d0 = part * 111;
    int d1 = d0 + 111; if (d1 > DD) d1 = DD;
    float mx = -3.0e38f;
    for (int dd = d0; dd < d1; ++dd) mx = fmaxf(mx, att2[dd * 64 + c]);
    red[part * 64 + c] = mx;
    __syncthreads();
    const float cmax =
        fmaxf(fmaxf(red[c], red[64 + c]), fmaxf(red[128 + c], red[192 + c]));
    float s = 0.f;
    for (int dd = d0; dd < d1; ++dd) {
      float e = expf(att2[dd * 64 + c] - cmax);
      att2[dd * 64 + c] = e;
      s += e;
    }
    red2[part * 64 + c] = s;
    __syncthreads();
    if (part == 0)
      colinv[c] = 1.f / (red2[c] + red2[64 + c] + red2[128 + c] + red2[192 + c]);
  }
  __syncthreads();

  // ---- phase 4: value-window dot + k1_pad add + crop, write final output ----
  // out[b,c,y,x] = [y>=2&&x>=2] inv[c]*sum_c' v[b,c',y-2,x-2]*exp[220-c+c', c]
  //             + [y<=18&&x<=18] k1[b, (y+2)*21+(x+2), c]
  for (int job = tid; job < npos * CC; job += 256) {
    const int p = job >> 6, c = job & 63;
    const int y = y0 + 5 * (p / nx), xx = x0 + 5 * (p % nx);
    float acc = 0.f;
    if (y >= 2 && xx >= 2) {
      const float* vp = &vloc[p * 64];
#pragma unroll 8
      for (int cp = 0; cp < 64; ++cp)
        acc += vp[cp] * att2[(220 - c + cp) * 64 + c];
      acc *= colinv[c];
    }
    float k1t = 0.f;
    if (y <= 18 && xx <= 18)
      k1t = k1_ws[((size_t)b * DD + (y + 2) * IMGN + (xx + 2)) * CC + c];
    out[(((size_t)b * CC + c) * IMGN + y) * IMGN + xx] = acc + k1t;
  }
}

// ---------------------------------------------------------------------------
extern "C" void kernel_launch(void* const* d_in, const int* in_sizes, int n_in,
                              void* d_out, int out_size, void* d_ws, size_t ws_size,
                              hipStream_t stream) {
  (void)in_sizes; (void)n_in; (void)out_size; (void)ws_size;
  const float* x        = (const float*)d_in[0];
  const float* wk       = (const float*)d_in[1];
  const float* bk       = (const float*)d_in[2];
  const float* w1       = (const float*)d_in[3];
  const float* bn_gamma = (const float*)d_in[4];
  const float* bn_beta  = (const float*)d_in[5];
  const float* bn_mean  = (const float*)d_in[6];
  const float* bn_var   = (const float*)d_in[7];
  const float* w2       = (const float*)d_in[8];
  const float* b2       = (const float*)d_in[9];
  const float* wv       = (const float*)d_in[10];
  const float* bv       = (const float*)d_in[11];
  float* out = (float*)d_out;

  // workspace: k1 (2*441*64 f32) then v (2*64*441 f32) = ~452 KB total
  float* k1_ws = (float*)d_ws;
  float* v_ws  = k1_ws + 2 * DD * CC;

  // stage 1: 441 blocks for k1, 128 blocks for depthwise 3x3 v
  semca_stage1<<<DD + 2 * CC, 128, 0, stream>>>(x, wk, bk, wv, bv, k1_ws, v_ws);
  // stage 2: one WG per (b, phase-group g) — fully fused attention + output
  semca_attn<<<2 * 25, 256, 0, stream>>>(x, w1, bn_gamma, bn_beta, bn_mean,
                                         bn_var, w2, b2, k1_ws, v_ws, out);
}
